// PAM_Module_36730560315895
// MI455X (gfx1250) — compile-verified, matrix operand order checked by reading, not run on hardware
//
#include <hip/hip_runtime.h>
#include <hip/hip_bf16.h>

typedef _Float16 half_t;
typedef half_t v16h __attribute__((ext_vector_type(16)));
typedef float  v8f  __attribute__((ext_vector_type(8)));
typedef int    v4i  __attribute__((ext_vector_type(4)));

typedef __attribute__((address_space(1))) v4i* gv4i_p;   // global int4*
typedef __attribute__((address_space(3))) v4i* lv4i_p;   // LDS int4*

union Frag {
    v16h v;
    uint4 q[2];
};

#if __has_builtin(__builtin_amdgcn_global_load_async_to_lds_b128)
#define HAS_ASYNC 1
#else
#define HAS_ASYNC 0
#endif

// 16-byte global -> LDS copy (async DMA on CDNA5, plain copy fallback)
__device__ __forceinline__ void async_cp16(void* ldsp, void* gp) {
#if HAS_ASYNC
    __builtin_amdgcn_global_load_async_to_lds_b128((gv4i_p)gp, (lv4i_p)ldsp, 0, 0);
#else
    *(uint4*)ldsp = *(const uint4*)gp;
#endif
}

#if HAS_ASYNC
#if __has_builtin(__builtin_amdgcn_s_wait_asynccnt)
#define WAIT_ASYNC(n) __builtin_amdgcn_s_wait_asynccnt(n)
#else
#define WAIT_ASYNC(n) asm volatile("s_wait_asynccnt %0" ::"i"(n) : "memory")
#endif
#else
#define WAIT_ASYNC(n) ((void)0)
#endif

// D = A*B + C, 16x16x32 f16 -> f32 accum
__device__ __forceinline__ v8f wmma32(v16h a, v16h b, v8f c) {
    return __builtin_amdgcn_wmma_f32_16x16x32_f16(false, a, false, b, (short)0, c, false, false);
}

// Load a 16x32 fragment (A-side: lane -> row m, per-lane k = kh..kh+7, kh+16..kh+23;
// B-side uses mirrored layout with lane -> column n).  ld = row stride
// in half_t, multiple of 8; rows 16-byte aligned.
__device__ __forceinline__ v16h load_frag(const half_t* base, int ld) {
    const int lane = threadIdx.x & 31;
    const int m  = lane & 15;
    const int kh = (lane >> 4) << 3;   // 0 or 8
    const half_t* p = base + m * ld + kh;
    Frag f;
    f.q[0] = *(const uint4*)(p);
    f.q[1] = *(const uint4*)(p + 16);
    return f.v;
}

__global__ void k_convert(const float* __restrict__ in, half_t* __restrict__ out, int n) {
    int i = blockIdx.x * blockDim.x + threadIdx.x;
    if (i < n) out[i] = (half_t)in[i];
}

// ---------------------------------------------------------------------------
// Kernel 1: QKV projection.  grid = B*32 blocks, 128 threads (4 waves).
// ---------------------------------------------------------------------------
#define XS_LD 520
__global__ void __launch_bounds__(128) __attribute__((amdgpu_waves_per_eu(1, 2))) k_qkv(
    const float* __restrict__ x,
    const half_t* __restrict__ Wq, const float* __restrict__ bq,
    const half_t* __restrict__ Wk, const float* __restrict__ bk,
    const half_t* __restrict__ Wv, const float* __restrict__ bv,
    half_t* __restrict__ Qo, half_t* __restrict__ Ko, half_t* __restrict__ Vo)
{
    extern __shared__ char smem[];
    half_t* xs = (half_t*)smem;              // [128 pixels][XS_LD] (c contiguous)

    const int b  = blockIdx.x >> 5;
    const int n0 = (blockIdx.x & 31) * 128;
    const int t  = threadIdx.x;
    const float* xb = x + (size_t)b * 512 * 4096;

    for (int idx = t; idx < 512 * 32; idx += 128) {
        int c = idx >> 5, n4 = (idx & 31) << 2;
        float4 f = *(const float4*)(xb + (size_t)c * 4096 + n0 + n4);
        xs[(n4 + 0) * XS_LD + c] = (half_t)f.x;
        xs[(n4 + 1) * XS_LD + c] = (half_t)f.y;
        xs[(n4 + 2) * XS_LD + c] = (half_t)f.z;
        xs[(n4 + 3) * XS_LD + c] = (half_t)f.w;
    }
    __syncthreads();

    const int w = t >> 5, lane = t & 31, ln = lane & 15, hi = lane >> 4;

    for (int mt = w; mt < 40; mt += 4) {
        const half_t* Wm; const float* bias; int mrow;
        if (mt < 4)       { Wm = Wq + (size_t)mt * 16 * 512;       bias = bq; mrow = mt * 16; }
        else if (mt < 8)  { Wm = Wk + (size_t)(mt - 4) * 16 * 512; bias = bk; mrow = (mt - 4) * 16; }
        else              { Wm = Wv + (size_t)(mt - 8) * 16 * 512; bias = bv; mrow = (mt - 8) * 16; }

        v8f acc[8];
        #pragma unroll
        for (int i = 0; i < 8; i++) acc[i] = (v8f){0.f, 0.f, 0.f, 0.f, 0.f, 0.f, 0.f, 0.f};

        for (int ks = 0; ks < 16; ks++) {
            v16h a = load_frag(Wm + ks * 32, 512);
            v16h bb[8];
            #pragma unroll
            for (int nt = 0; nt < 8; nt++)
                bb[nt] = load_frag(xs + (nt * 16) * XS_LD + ks * 32, XS_LD);
            #pragma unroll
            for (int nt = 0; nt < 8; nt++)
                acc[nt] = wmma32(a, bb[nt], acc[nt]);
        }

        #pragma unroll
        for (int nt = 0; nt < 8; nt++) {
            int pix = n0 + nt * 16 + ln;
            #pragma unroll
            for (int r = 0; r < 8; r++) {
                int rr = r + hi * 8;
                float val = acc[nt][r] + bias[mrow + rr];
                half_t hv = (half_t)val;
                if (mt < 4)
                    Qo[((size_t)b * 4096 + pix) * 64 + mrow + rr] = hv;
                else if (mt < 8)
                    Ko[((size_t)b * 4096 + pix) * 64 + mrow + rr] = hv;
                else
                    Vo[((size_t)b * 512 + mrow + rr) * 4096 + pix] = hv;
            }
        }
    }
}

// ---------------------------------------------------------------------------
// Kernel 2: fused attention. grid = B*256 (16 query rows each), 128 threads.
// pass1: online softmax stats; pass2: P (f16) resident in LDS;
// pass3: PV GEMM with async double-buffered V staging.
// ---------------------------------------------------------------------------
#define PLD 4104
__global__ void __launch_bounds__(128) __attribute__((amdgpu_waves_per_eu(1, 2))) k_attn(
    const half_t* __restrict__ Qg, const half_t* __restrict__ Kg,
    const half_t* __restrict__ Vg,
    const float* __restrict__ x, const float* __restrict__ gamma,
    half_t* __restrict__ Yo)
{
    extern __shared__ char smem[];
    half_t* Pp = (half_t*)smem;              // 16 x PLD
    half_t* Qs = Pp + 16 * PLD;              // 16 x 72
    half_t* Vs = Qs + 16 * 72;               // 2 x (512 x 32) double buffer
    float*  Ms = (float*)(Vs + 2 * 512 * 32);
    float*  Ss = Ms + 16;
    float*  Mw = Ss + 16;                    // 4 x 16
    float*  Sw = Mw + 64;                    // 4 x 16
    float*  Scr = Sw + 64;                   // 4 x (16 x 17)

    const int b  = blockIdx.x >> 8;
    const int i0 = (blockIdx.x & 255) * 16;
    const int t = threadIdx.x, w = t >> 5, lane = t & 31, ln = lane & 15, hi = lane >> 4;
    const half_t* Qb = Qg + ((size_t)b * 4096 + i0) * 64;
    const half_t* Kb = Kg + (size_t)b * 4096 * 64;
    const half_t* Vb = Vg + (size_t)b * 512 * 4096;
    float* scr = Scr + w * 272;

    for (int idx = t; idx < 16 * 64; idx += 128) {
        int r = idx >> 6, c = idx & 63;
        Qs[r * 72 + c] = Qb[r * 64 + c];
    }
    __syncthreads();

    const v16h aQ0 = load_frag(Qs, 72);
    const v16h aQ1 = load_frag(Qs + 32, 72);
    const v8f vzero = (v8f){0.f, 0.f, 0.f, 0.f, 0.f, 0.f, 0.f, 0.f};

    // ---- pass 1: per-row running max / sum(exp) ----
    float rm = -3.0e38f, rs = 0.f;
    for (int jt = w; jt < 256; jt += 4) {
        int j0 = jt * 16;
        __builtin_prefetch(Kb + (size_t)(j0 + 64) * 64, 0, 1);
        v16h b0 = load_frag(Kb + (size_t)j0 * 64, 64);
        v16h b1 = load_frag(Kb + (size_t)j0 * 64 + 32, 64);
        v8f e = vzero;
        e = wmma32(aQ0, b0, e);
        e = wmma32(aQ1, b1, e);
        #pragma unroll
        for (int r = 0; r < 8; r++) scr[(r + hi * 8) * 17 + ln] = e[r];
        asm volatile("s_wait_dscnt 0x0" ::: "memory");
        float vals[16];
        float tmax = -3.0e38f;
        #pragma unroll
        for (int c = 0; c < 16; c++) { vals[c] = scr[ln * 17 + c]; tmax = fmaxf(tmax, vals[c]); }
        float nm = fmaxf(rm, tmax);
        float ssum = 0.f;
        #pragma unroll
        for (int c = 0; c < 16; c++) ssum += __expf(vals[c] - nm);
        rs = rs * __expf(rm - nm) + ssum;
        rm = nm;
        asm volatile("s_wait_dscnt 0x0" ::: "memory");
    }
    if (lane < 16) { Mw[w * 16 + lane] = rm; Sw[w * 16 + lane] = rs; }
    __syncthreads();
    if (t < 16) {
        float gm = Mw[t];
        for (int ww = 1; ww < 4; ww++) gm = fmaxf(gm, Mw[ww * 16 + t]);
        float gs = 0.f;
        for (int ww = 0; ww < 4; ww++) gs += Sw[ww * 16 + t] * __expf(Mw[ww * 16 + t] - gm);
        Ms[t] = gm;
        Ss[t] = 1.0f / gs;
    }
    __syncthreads();

    // ---- pass 2: recompute energy, write normalized P to LDS ----
    for (int jt = w; jt < 256; jt += 4) {
        int j0 = jt * 16;
        __builtin_prefetch(Kb + (size_t)(j0 + 64) * 64, 0, 1);
        v16h b0 = load_frag(Kb + (size_t)j0 * 64, 64);
        v16h b1 = load_frag(Kb + (size_t)j0 * 64 + 32, 64);
        v8f e = vzero;
        e = wmma32(aQ0, b0, e);
        e = wmma32(aQ1, b1, e);
        #pragma unroll
        for (int r = 0; r < 8; r++) {
            int row = r + hi * 8;
            float p = __expf(e[r] - Ms[row]) * Ss[row];
            Pp[row * PLD + j0 + ln] = (half_t)p;
        }
    }
    __syncthreads();

    // ---- pass 3: PV GEMM, wave w owns v in [w*128, w*128+128) ----
    v8f acc[8];
    #pragma unroll
    for (int i = 0; i < 8; i++) acc[i] = vzero;
    const int vbase = w * 128;

    // stage j-slice [j0, j0+32) of V (all 512 rows) into buffer `buf`
    auto stage_v = [&](half_t* buf, int j0) {
        for (int row = t; row < 512; row += 128) {
            char* d = (char*)(buf + row * 32);
            char* s = (char*)(Vb + (size_t)row * 4096 + j0);
            async_cp16(d,      s);
            async_cp16(d + 16, s + 16);
            async_cp16(d + 32, s + 32);
            async_cp16(d + 48, s + 48);
        }
    };

    stage_v(Vs, 0);                        // 16 async ops per thread
    for (int js = 0; js < 128; js++) {
        half_t* cur = Vs + (js & 1) * 512 * 32;
        half_t* nxt = Vs + ((js + 1) & 1) * 512 * 32;
        __syncthreads();                   // all readers done with nxt buffer
        if (js < 127) {
            stage_v(nxt, (js + 1) * 32);   // overlaps with compute below
            WAIT_ASYNC(16);                // previous stage (cur) complete
        } else {
            WAIT_ASYNC(0);
        }
        __syncthreads();                   // cur staged by all threads

        v16h aP = load_frag(Pp + js * 32, PLD);
        v16h bf[8];
        #pragma unroll
        for (int nt = 0; nt < 8; nt++)
            bf[nt] = load_frag(cur + (size_t)(vbase + nt * 16) * 32, 32);
        #pragma unroll
        for (int nt = 0; nt < 8; nt++)
            acc[nt] = wmma32(aP, bf[nt], acc[nt]);
    }

    // ---- epilogue: gamma*out + x, LDS tile transpose, coalesced store ----
    const float g = gamma[0];
    const float* xb = x + (size_t)b * 512 * 4096;
    for (int nt = 0; nt < 8; nt++) {
        #pragma unroll
        for (int r = 0; r < 8; r++) scr[(r + hi * 8) * 17 + ln] = acc[nt][r];
        asm volatile("s_wait_dscnt 0x0" ::: "memory");
        for (int e = lane; e < 256; e += 32) {
            int vv = e >> 4, ii = e & 15;
            int c = vbase + nt * 16 + vv;
            int pix = i0 + ii;
            float val = g * scr[ii * 17 + vv] + xb[(size_t)c * 4096 + pix];
            Yo[((size_t)b * 512 + c) * 4096 + pix] = (half_t)val;
        }
        asm volatile("s_wait_dscnt 0x0" ::: "memory");
    }
}

// ---------------------------------------------------------------------------
// Kernel 3: 1x1 conv (Wf) + BatchNorm + ReLU.
// ---------------------------------------------------------------------------
__global__ void __launch_bounds__(128) __attribute__((amdgpu_waves_per_eu(1, 2))) k_fuse(
    const half_t* __restrict__ Y, const half_t* __restrict__ Wf,
    const float* __restrict__ bn_w, const float* __restrict__ bn_b,
    const float* __restrict__ bn_mean, const float* __restrict__ bn_var,
    float* __restrict__ out)
{
    extern __shared__ char smem[];
    half_t* ys = (half_t*)smem;              // [128][XS_LD]
    const int b  = blockIdx.x >> 5;
    const int n0 = (blockIdx.x & 31) * 128;
    const int t  = threadIdx.x;
    const half_t* yb = Y + (size_t)b * 512 * 4096;

    for (int idx = t; idx < 512 * 16; idx += 128) {
        int c = idx >> 4, n8 = (idx & 15) << 3;
        union { uint4 q; half_t h[8]; } u;
        u.q = *(const uint4*)(yb + (size_t)c * 4096 + n0 + n8);
        #pragma unroll
        for (int j = 0; j < 8; j++) ys[(n8 + j) * XS_LD + c] = u.h[j];
    }
    __syncthreads();

    const int w = t >> 5, lane = t & 31, ln = lane & 15, hi = lane >> 4;
    for (int mt = w; mt < 32; mt += 4) {
        int mrow = mt * 16;
        v8f acc[8];
        #pragma unroll
        for (int i = 0; i < 8; i++) acc[i] = (v8f){0.f, 0.f, 0.f, 0.f, 0.f, 0.f, 0.f, 0.f};
        for (int ks = 0; ks < 16; ks++) {
            v16h a = load_frag(Wf + (size_t)mrow * 512 + ks * 32, 512);
            v16h bb[8];
            #pragma unroll
            for (int nt = 0; nt < 8; nt++)
                bb[nt] = load_frag(ys + (nt * 16) * XS_LD + ks * 32, XS_LD);
            #pragma unroll
            for (int nt = 0; nt < 8; nt++)
                acc[nt] = wmma32(a, bb[nt], acc[nt]);
        }
        #pragma unroll
        for (int nt = 0; nt < 8; nt++) {
            #pragma unroll
            for (int r = 0; r < 8; r++) {
                int o = mrow + r + hi * 8;
                float scale = bn_w[o] * rsqrtf(bn_var[o] + 1e-5f);
                float shift = bn_b[o] - bn_mean[o] * scale;
                float val = fmaxf(acc[nt][r] * scale + shift, 0.f);
                out[((size_t)b * 512 + o) * 4096 + n0 + nt * 16 + ln] = val;
            }
        }
    }
}

extern "C" void kernel_launch(void* const* d_in, const int* in_sizes, int n_in,
                              void* d_out, int out_size, void* d_ws, size_t ws_size,
                              hipStream_t stream) {
    const float* x     = (const float*)d_in[0];
    const float* Wq    = (const float*)d_in[1];
    const float* bq    = (const float*)d_in[2];
    const float* Wk    = (const float*)d_in[3];
    const float* bk    = (const float*)d_in[4];
    const float* Wv    = (const float*)d_in[5];
    const float* bv    = (const float*)d_in[6];
    const float* gamma = (const float*)d_in[7];
    const float* Wf    = (const float*)d_in[8];
    const float* bn_w  = (const float*)d_in[9];
    const float* bn_b  = (const float*)d_in[10];
    const float* bn_m  = (const float*)d_in[11];
    const float* bn_v  = (const float*)d_in[12];

    char* ws = (char*)d_ws;
    half_t* WqB = (half_t*)(ws);                 // 64*512
    half_t* WkB = (half_t*)(ws + 65536);         // 64*512
    half_t* WvB = (half_t*)(ws + 131072);        // 512*512
    half_t* WfB = (half_t*)(ws + 655360);        // 512*512
    half_t* Qb  = (half_t*)(ws + 1179648);       // 8*4096*64
    half_t* Kb  = (half_t*)(ws + 5373952);       // 8*4096*64
    half_t* Vb  = (half_t*)(ws + 9568256);       // 8*512*4096
    half_t* Yb  = (half_t*)(ws + 43122688);      // 8*512*4096

    k_convert<<<128,  256, 0, stream>>>(Wq, WqB, 64 * 512);
    k_convert<<<128,  256, 0, stream>>>(Wk, WkB, 64 * 512);
    k_convert<<<1024, 256, 0, stream>>>(Wv, WvB, 512 * 512);
    k_convert<<<1024, 256, 0, stream>>>(Wf, WfB, 512 * 512);

    const int qkvLds  = 128 * XS_LD * 2;                                      // 133120
    const int attnLds = (16 * PLD + 16 * 72 + 2 * 512 * 32) * 2 + 1248 * 4;   // 204160

    (void)hipFuncSetAttribute((const void*)k_qkv,  hipFuncAttributeMaxDynamicSharedMemorySize, qkvLds);
    (void)hipFuncSetAttribute((const void*)k_attn, hipFuncAttributeMaxDynamicSharedMemorySize, attnLds);
    (void)hipFuncSetAttribute((const void*)k_fuse, hipFuncAttributeMaxDynamicSharedMemorySize, qkvLds);

    k_qkv<<<256, 128, qkvLds, stream>>>(x, WqB, bq, WkB, bk, WvB, bv, Qb, Kb, Vb);
    k_attn<<<2048, 128, attnLds, stream>>>(Qb, Kb, Vb, x, gamma, Yb);
    k_fuse<<<256, 128, qkvLds, stream>>>(Yb, WfB, bn_w, bn_b, bn_m, bn_v, (float*)d_out);
}
